// EnhancedHierarchicalDeltaNet_70205535420720
// MI455X (gfx1250) — matrix operand order repaired and checked by reading, not compiled, not run on hardware
//
#include <hip/hip_runtime.h>
#include <hip/hip_bf16.h>
#include <cstdint>

// Problem constants (match reference)
#define BB   4
#define LL   4096
#define DD   1024
#define HH   4
#define DKK  256
#define CC   64
#define NCC  64
#define KSS  4
#define FHH  128      // DK/2
#define VBLK 8        // v-dim splits in scan kernel
#define VW   32       // v width per scan block (DKK / VBLK)

typedef __attribute__((ext_vector_type(16))) __bf16 v16bf;
typedef __attribute__((ext_vector_type(8)))  float  v8f;

__device__ __forceinline__ float sigf(float x) { return 1.f / (1.f + __expf(-x)); }

__device__ __forceinline__ float waveSum(float v) {
#pragma unroll
  for (int off = 16; off > 0; off >>= 1) v += __shfl_xor(v, off, 32);
  return v;
}

__device__ __forceinline__ uint4 pack8(float4 a, float4 b) {
  union { uint4 u; __bf16 h[8]; } r;
  r.h[0] = (__bf16)a.x; r.h[1] = (__bf16)a.y; r.h[2] = (__bf16)a.z; r.h[3] = (__bf16)a.w;
  r.h[4] = (__bf16)b.x; r.h[5] = (__bf16)b.y; r.h[6] = (__bf16)b.z; r.h[7] = (__bf16)b.w;
  return r.u;
}
__device__ __forceinline__ void unpack8(uint4 u, float4& a, float4& b) {
  union { uint4 u4; __bf16 h[8]; } r;
  r.u4 = u;
  a = make_float4((float)r.h[0], (float)r.h[1], (float)r.h[2], (float)r.h[3]);
  b = make_float4((float)r.h[4], (float)r.h[5], (float)r.h[6], (float)r.h[7]);
}
__device__ __forceinline__ float4 scale4(float4 a, float s) {
  return make_float4(a.x * s, a.y * s, a.z * s, a.w * s);
}
__device__ __forceinline__ float4 mul4(float4 a, float4 b) {
  return make_float4(a.x * b.x, a.y * b.y, a.z * b.z, a.w * b.w);
}
__device__ __forceinline__ float dot4(float4 a, float4 b) {
  return a.x * b.x + a.y * b.y + a.z * b.z + a.w * b.w;
}
__device__ __forceinline__ float4 sig4(float4 a) {
  return make_float4(sigf(a.x), sigf(a.y), sigf(a.z), sigf(a.w));
}

__device__ __forceinline__ v8f wmma_bf(v16bf a, v16bf b, v8f c) {
  return __builtin_amdgcn_wmma_f32_16x16x32_bf16(false, a, false, b, (short)0, c, false, false);
}

// ---- wave32 WMMA fragment loaders (ISA 7.12.2 layouts) ----
__device__ __forceinline__ v16bf load_a_rm(const __bf16* p, int ld) {
  int lane = threadIdx.x & 31;
  const __bf16* row = p + (lane & 15) * ld + (lane >> 4) * 8;
  v16bf a;
#pragma unroll
  for (int j = 0; j < 8; ++j) { a[j] = row[j]; a[8 + j] = row[16 + j]; }
  return a;
}
__device__ __forceinline__ v16bf load_a_cm(const __bf16* p, int ld) {
  int lane = threadIdx.x & 31;
  int m = lane & 15, g8 = (lane >> 4) * 8;
  v16bf a;
#pragma unroll
  for (int j = 0; j < 8; ++j) {
    a[j]     = p[(g8 + j) * ld + m];
    a[8 + j] = p[(g8 + 16 + j) * ld + m];
  }
  return a;
}
__device__ __forceinline__ v16bf load_b_nm(const __bf16* p, int ld) {
  int lane = threadIdx.x & 31;
  const __bf16* col = p + (lane & 15) * ld + (lane >> 4) * 8;
  v16bf b;
#pragma unroll
  for (int t = 0; t < 16; ++t) b[t] = col[t < 8 ? t : t + 8];
  return b;
}
__device__ __forceinline__ v16bf load_b_km(const __bf16* p, int ld) {
  int lane = threadIdx.x & 31;
  int n = lane & 15, g8 = (lane >> 4) * 8;
  v16bf b;
#pragma unroll
  for (int t = 0; t < 16; ++t) b[t] = p[(g8 + (t < 8 ? t : t + 8)) * ld + n];
  return b;
}
__device__ __forceinline__ void store_tile(float* p, int ld, v8f acc) {
  int lane = threadIdx.x & 31;
  int n = lane & 15, m0 = (lane >> 4) * 8;
#pragma unroll
  for (int r = 0; r < 8; ++r) p[(size_t)(m0 + r) * ld + n] = acc[r];
}
__device__ __forceinline__ void store_tile_bf16(__bf16* p, int ld, v8f acc) {
  int lane = threadIdx.x & 31;
  int n = lane & 15, m0 = (lane >> 4) * 8;
#pragma unroll
  for (int r = 0; r < 8; ++r) p[(size_t)(m0 + r) * ld + n] = (__bf16)acc[r];
}

// =====================================================================
// Kernel 0a: flat f32 -> bf16 convert
// =====================================================================
__global__ __launch_bounds__(256) void cvt_bf16_kernel(const float* __restrict__ in,
                                                       __bf16* __restrict__ out, size_t n8) {
  size_t i = (size_t)blockIdx.x * 256 + threadIdx.x;
  if (i >= n8) return;
  const float4* p = (const float4*)in + i * 2;
  float4 a = p[0], b = p[1];
  *((uint4*)out + i) = pack8(a, b);
}

// =====================================================================
// Kernel 0b: W(KxN,f32) -> WT(NxK,bf16) transpose via 64x64 LDS tile
// =====================================================================
__global__ __launch_bounds__(256) void cvtT_bf16_kernel(const float* __restrict__ W,
                                                        __bf16* __restrict__ WT, int K, int N) {
  __shared__ __bf16 sT[64 * 72];
  int bk = blockIdx.x * 64, bn = blockIdx.y * 64;
  int tid = threadIdx.x;
#pragma unroll
  for (int i = 0; i < 16; ++i) {
    int idx = tid + i * 256;
    int kr = idx >> 6, nc = idx & 63;
    sT[nc * 72 + kr] = (__bf16)W[(size_t)(bk + kr) * N + bn + nc];
  }
  __syncthreads();
#pragma unroll
  for (int i = 0; i < 2; ++i) {
    int j = tid + i * 256;
    int nr = j >> 3, c8 = j & 7;
    uint4 v = *(const uint4*)(&sT[nr * 72 + c8 * 8]);
    *(uint4*)(WT + (size_t)(bn + nr) * K + bk + c8 * 8) = v;
  }
}

// =====================================================================
// Kernel 1: Y(MxN) = X(MxK,bf16) @ WT(NxK,bf16)^T; output f32 or bf16.
// 128x128 block, BK=32, ping-pong LDS, 8 waves as 4(M)x2(N), 32x64/wave.
// =====================================================================
#define GBM 128
#define GBN 128
#define GBK 32
template <bool BF16OUT>
__global__ __launch_bounds__(256)
__attribute__((amdgpu_waves_per_eu(2)))
void gemm_bf16_kernel(const __bf16* __restrict__ X, const __bf16* __restrict__ WT,
                      void* __restrict__ Yv, int M, int N, int K) {
  __shared__ __bf16 sA[2][GBM * GBK];
  __shared__ __bf16 sB[2][GBN * GBK];        // n-major [n][k]
  int bm = blockIdx.x * GBM, bn = blockIdx.y * GBN;
  int tid = threadIdx.x, wave = tid >> 5;
  int wm = wave >> 1, wn = wave & 1;         // 4 x 2 wave grid
  uint4 rx[2], rw[2];
  auto fetch = [&](int k0) {
#pragma unroll
    for (int i = 0; i < 2; ++i) {
      int u = tid + i * 256;                 // 512 uint4 per tile
      int row = u >> 2, c4 = u & 3;
      rx[i] = *(const uint4*)(X + (size_t)(bm + row) * K + k0 + c4 * 8);
      rw[i] = *(const uint4*)(WT + (size_t)(bn + row) * K + k0 + c4 * 8);
    }
  };
  auto stash = [&](int b) {
#pragma unroll
    for (int i = 0; i < 2; ++i) {
      int u = tid + i * 256;
      int row = u >> 2, c4 = u & 3;
      *(uint4*)(&sA[b][row * GBK + c4 * 8]) = rx[i];
      *(uint4*)(&sB[b][row * GBK + c4 * 8]) = rw[i];
    }
  };
  v8f acc[8] = {};
  fetch(0);
  stash(0);
  __syncthreads();
  int buf = 0;
  for (int k0 = 0; k0 < K; k0 += GBK) {
    bool last = (k0 + GBK >= K);
    if (!last) fetch(k0 + GBK);
    v16bf a0 = load_a_rm(&sA[buf][(wm * 32 + 0) * GBK], GBK);
    v16bf a1 = load_a_rm(&sA[buf][(wm * 32 + 16) * GBK], GBK);
    v16bf b0 = load_b_nm(&sB[buf][(wn * 64 + 0) * GBK], GBK);
    v16bf b1 = load_b_nm(&sB[buf][(wn * 64 + 16) * GBK], GBK);
    v16bf b2 = load_b_nm(&sB[buf][(wn * 64 + 32) * GBK], GBK);
    v16bf b3 = load_b_nm(&sB[buf][(wn * 64 + 48) * GBK], GBK);
    acc[0] = wmma_bf(a0, b0, acc[0]);
    acc[1] = wmma_bf(a0, b1, acc[1]);
    acc[2] = wmma_bf(a0, b2, acc[2]);
    acc[3] = wmma_bf(a0, b3, acc[3]);
    acc[4] = wmma_bf(a1, b0, acc[4]);
    acc[5] = wmma_bf(a1, b1, acc[5]);
    acc[6] = wmma_bf(a1, b2, acc[6]);
    acc[7] = wmma_bf(a1, b3, acc[7]);
    if (!last) { stash(buf ^ 1); __syncthreads(); buf ^= 1; }
  }
#pragma unroll
  for (int i = 0; i < 2; ++i)
#pragma unroll
    for (int j = 0; j < 4; ++j) {
      size_t off = (size_t)(bm + wm * 32 + i * 16) * N + bn + wn * 64 + j * 16;
      if constexpr (BF16OUT)
        store_tile_bf16((__bf16*)Yv + off, N, acc[i * 4 + j]);
      else
        store_tile((float*)Yv + off, N, acc[i * 4 + j]);
    }
}

// =====================================================================
// Kernel 2: beta = sigmoid(x @ Wb) — one wave per row, 4 heads at once
// =====================================================================
__global__ __launch_bounds__(256) void beta_kernel(const float* __restrict__ X,
                                                   const float* __restrict__ Wb,
                                                   float* __restrict__ beta) {
  int row = blockIdx.x * 8 + (threadIdx.x >> 5);
  int lane = threadIdx.x & 31;
  const float* xr = X + (size_t)row * DD;
  float4 acc = make_float4(0.f, 0.f, 0.f, 0.f);
  for (int k = lane * 4; k < DD; k += 128) {
    float4 xv = *(const float4*)(xr + k);
    float4 w0 = *(const float4*)(Wb + (size_t)(k + 0) * HH);
    float4 w1 = *(const float4*)(Wb + (size_t)(k + 1) * HH);
    float4 w2 = *(const float4*)(Wb + (size_t)(k + 2) * HH);
    float4 w3 = *(const float4*)(Wb + (size_t)(k + 3) * HH);
    acc.x += xv.x * w0.x + xv.y * w1.x + xv.z * w2.x + xv.w * w3.x;
    acc.y += xv.x * w0.y + xv.y * w1.y + xv.z * w2.y + xv.w * w3.y;
    acc.z += xv.x * w0.z + xv.y * w1.z + xv.z * w2.z + xv.w * w3.z;
    acc.w += xv.x * w0.w + xv.y * w1.w + xv.z * w2.w + xv.w * w3.w;
  }
  acc.x = waveSum(acc.x); acc.y = waveSum(acc.y);
  acc.z = waveSum(acc.z); acc.w = waveSum(acc.w);
  if (lane == 0) *(float4*)(beta + (size_t)row * HH) = sig4(acc);
}

// =====================================================================
// Kernel 3: depthwise causal conv (KS=4, pad 2/1) + SiLU, bf16 in/out
// =====================================================================
__global__ __launch_bounds__(256) void conv_silu_kernel(const __bf16* __restrict__ in,
                                                        const float* __restrict__ cw,
                                                        __bf16* __restrict__ out) {
  size_t i4 = (size_t)blockIdx.x * 256 + threadIdx.x;
  if (i4 >= (size_t)BB * LL * DD / 4) return;
  size_t idx = i4 * 4;
  int d = (int)(idx % DD);
  int l = (int)((idx / DD) % LL);
  float c0[4], c1[4], c2[4], c3[4];
  *(float4*)c0 = *(const float4*)(cw + (size_t)(d + 0) * KSS);
  *(float4*)c1 = *(const float4*)(cw + (size_t)(d + 1) * KSS);
  *(float4*)c2 = *(const float4*)(cw + (size_t)(d + 2) * KSS);
  *(float4*)c3 = *(const float4*)(cw + (size_t)(d + 3) * KSS);
  float4 acc = make_float4(0.f, 0.f, 0.f, 0.f);
#pragma unroll
  for (int t = 0; t < KSS; ++t) {
    int ls = l - 2 + t;
    if (ls >= 0 && ls < LL) {
      union { uint2 u; __bf16 h[4]; } r;
      r.u = *(const uint2*)(in + (size_t)((long)idx + (long)(ls - l) * DD));
      acc.x += (float)r.h[0] * c0[t];
      acc.y += (float)r.h[1] * c1[t];
      acc.z += (float)r.h[2] * c2[t];
      acc.w += (float)r.h[3] * c3[t];
    }
  }
  float4 y = mul4(acc, sig4(acc));
  union { uint2 u; __bf16 h[4]; } o;
  o.h[0] = (__bf16)y.x; o.h[1] = (__bf16)y.y; o.h[2] = (__bf16)y.z; o.h[3] = (__bf16)y.w;
  *(uint2*)(out + idx) = o.u;
}

// =====================================================================
// Kernel 4: per-(b,h,chunk) delta-rule prep (bf16 q/k/v inputs).
// =====================================================================
__global__ __launch_bounds__(256)
__attribute__((amdgpu_waves_per_eu(2)))
void chunk_kernel(
    const __bf16* __restrict__ qg, const __bf16* __restrict__ kg, const __bf16* __restrict__ vg,
    const float* __restrict__ betag,
    const float* __restrict__ Wbil, const float* __restrict__ temp,
    const float* __restrict__ fw1, const float* __restrict__ fb1,
    const float* __restrict__ fw2, const float* __restrict__ fb2,
    __bf16* __restrict__ qbg, __bf16* __restrict__ kbg, __bf16* __restrict__ wbg,
    float* __restrict__ ug, float* __restrict__ psig) {
  extern __shared__ __align__(16) char smem[];
  float* Am    = (float*)smem;           // [64][65]
  float* Tm    = Am + 64 * 65;           // [64][65]
  float* kmean = Tm + 64 * 65;           // [256]
  float* umean = kmean + 256;            // [256]
  float* h1    = umean + 256;            // [128]
  float* red   = h1 + 128;               // [16]
  __bf16* kbf = (__bf16*)(red + 16);     // [64][256]
  __bf16* kbb = kbf + 64 * 256;          // [64][256]
  __bf16* vbb = kbb + 64 * 256;          // [64][256]
  __bf16* ubf = vbb + 64 * 256;          // [64][256]
  __bf16* Tbf = ubf + 64 * 256;          // [64][64]

  int n = blockIdx.x % NCC;
  int h = (blockIdx.x / NCC) % HH;
  int b = blockIdx.x / (NCC * HH);
  int tid = threadIdx.x, lane = tid & 31, wave = tid >> 5;
  int nn = lane & 15, gg = lane >> 4;
  size_t gbase = ((size_t)b * LL + (size_t)n * CC) * DD + (size_t)h * DKK;

  // --- l2norm q,k; build k/k_beta/v_beta LDS; write bf16 q,k globals ---
  for (int rr = 0; rr < 8; ++rr) {
    int c = wave * 8 + rr;
    size_t rb = gbase + (size_t)c * DD;
    int dk0 = lane * 8;
    float4 q0, q1, k0, k1, v0, v1;
    unpack8(*(const uint4*)(qg + rb + dk0), q0, q1);
    unpack8(*(const uint4*)(kg + rb + dk0), k0, k1);
    unpack8(*(const uint4*)(vg + rb + dk0), v0, v1);
    float ssq = waveSum(dot4(q0, q0) + dot4(q1, q1));
    float ssk = waveSum(dot4(k0, k0) + dot4(k1, k1));
    float iq = rsqrtf(ssq + 1e-6f), ik = rsqrtf(ssk + 1e-6f);
    float bet = betag[((size_t)b * LL + (size_t)n * CC + c) * HH + h];
    float4 kn0 = scale4(k0, ik), kn1 = scale4(k1, ik);
    uint4 k8 = pack8(kn0, kn1);
    *(uint4*)(qbg + rb + dk0) = pack8(scale4(q0, iq), scale4(q1, iq));
    *(uint4*)(kbg + rb + dk0) = k8;
    *(uint4*)(&kbf[c * 256 + dk0]) = k8;
    *(uint4*)(&kbb[c * 256 + dk0]) = pack8(scale4(kn0, bet), scale4(kn1, bet));
    *(uint4*)(&vbb[c * 256 + dk0]) = pack8(scale4(v0, bet), scale4(v1, bet));
  }
  __syncthreads();

  // --- A = (k_beta @ k^T) * strict ---
#pragma unroll
  for (int t2 = 0; t2 < 2; ++t2) {
    int ti = wave * 2 + t2, tm = ti >> 2, tn = ti & 3;
    v8f acc = {};
    for (int kk = 0; kk < 256; kk += 32) {
      v16bf a   = load_a_rm(kbb + tm * 16 * 256 + kk, 256);
      v16bf bfr = load_b_nm(kbf + tn * 16 * 256 + kk, 256);
      acc = wmma_bf(a, bfr, acc);
    }
#pragma unroll
    for (int r = 0; r < 8; ++r) {
      int gm = tm * 16 + r + 8 * gg, gn = tn * 16 + nn;
      Am[gm * 65 + gn] = (gm > gn) ? acc[r] : 0.f;
    }
  }
  __syncthreads();

  // --- T = (I+A)^-1 via forward substitution (unit lower triangular) ---
  if (tid < 64) Tm[tid] = (tid == 0) ? 1.f : 0.f;
  __syncthreads();
  for (int i = 1; i < 64; ++i) {
    if (tid < 64) {
      float s = (tid == i) ? 1.f : 0.f;
      for (int t = 0; t < i; ++t) s -= Am[i * 65 + t] * Tm[t * 65 + tid];
      Tm[i * 65 + tid] = s;
    }
    __syncthreads();
  }
  for (int i = tid; i < 64 * 64; i += 256) Tbf[i] = (__bf16)Tm[(i >> 6) * 65 + (i & 63)];
  __syncthreads();

  // --- w = T @ k_beta (bf16 out) ; u = T @ v_beta (f32 + LDS bf16) ---
#pragma unroll
  for (int t8 = 0; t8 < 8; ++t8) {
    int ti = wave * 8 + t8, tm = ti >> 4, tn = ti & 15;
    v8f accw = {}, accu = {};
    for (int kk = 0; kk < 64; kk += 32) {
      v16bf a  = load_a_rm(Tbf + tm * 16 * 64 + kk, 64);
      v16bf bw = load_b_km(kbb + kk * 256 + tn * 16, 256);
      v16bf bu = load_b_km(vbb + kk * 256 + tn * 16, 256);
      accw = wmma_bf(a, bw, accw);
      accu = wmma_bf(a, bu, accu);
    }
    store_tile_bf16(wbg + gbase + (size_t)(tm * 16) * DD + tn * 16, DD, accw);
#pragma unroll
    for (int r = 0; r < 8; ++r) {
      int c = tm * 16 + r + 8 * gg, col = tn * 16 + nn;
      float uv = accu[r];
      ug[gbase + (size_t)c * DD + col] = uv;
      ubf[c * 256 + col] = (__bf16)uv;
    }
  }
  __syncthreads();

  // --- column means of k, u (flux inputs) ---
  {
    float sk = 0.f, su = 0.f;
    for (int c = 0; c < 64; ++c) {
      sk += (float)kbf[c * 256 + tid];
      su += (float)ubf[c * 256 + tid];
    }
    kmean[tid] = sk * (1.f / 64.f);
    umean[tid] = su * (1.f / 64.f);
  }

  // --- dot = sum_{d,v} Wbil[h][d][v] * (k^T u)[d][v] ---
  float dotp = 0.f;
  const float* Wbh = Wbil + (size_t)h * DKK * DKK;
  for (int t32 = 0; t32 < 32; ++t32) {
    int ti = wave * 32 + t32, tm = ti >> 4, tn = ti & 15;
    v8f acc = {};
    for (int kk = 0; kk < 64; kk += 32) {
      v16bf a   = load_a_cm(kbf + kk * 256 + tm * 16, 256);
      v16bf bfr = load_b_km(ubf + kk * 256 + tn * 16, 256);
      acc = wmma_bf(a, bfr, acc);
    }
#pragma unroll
    for (int r = 0; r < 8; ++r) {
      int dd = tm * 16 + r + 8 * gg, vv = tn * 16 + nn;
      dotp += acc[r] * Wbh[(size_t)dd * DKK + vv];
    }
  }
  dotp = waveSum(dotp);
  if (lane == 0) red[wave] = dotp;
  __syncthreads();
  if (tid == 0) {
    float s = 0.f;
    for (int i = 0; i < 8; ++i) s += red[i];
    red[8] = s;
  }
  __syncthreads();
  float aattn = red[8] / (64.f * temp[h]);

  // --- flux MLP -> psi ---
  if (tid < FHH) {
    float acc = fb1[tid];
    for (int i = 0; i < 256; ++i) acc += kmean[i] * fw1[i * FHH + tid];
    for (int i = 0; i < 256; ++i) acc += umean[i] * fw1[(256 + i) * FHH + tid];
    acc += aattn * fw1[512 * FHH + tid];
    float hv = acc * sigf(acc);
    h1[tid] = hv * fw2[tid];
  }
  __syncthreads();
  if (tid == 0) {
    float z = fb2[0];
    for (int j = 0; j < FHH; ++j) z += h1[j];
    float ps = sigf(z);
    psig[(size_t)(b * HH + h) * NCC + n] = fminf(0.99f, fmaxf(0.01f, ps));
  }
}

// =====================================================================
// Kernel 5: sequential scan over 64 chunks. One WG per (b,h,v-slice).
// =====================================================================
__global__ __launch_bounds__(256)
__attribute__((amdgpu_waves_per_eu(2)))
void scan_kernel(
    const __bf16* __restrict__ qgb, const __bf16* __restrict__ kgb,
    const __bf16* __restrict__ wgb, const float* __restrict__ ug,
    const float* __restrict__ psig, const float* __restrict__ lamf,
    const float* __restrict__ lams, float* __restrict__ og) {
  extern __shared__ __align__(16) char smem[];
  float* Sf = (float*)smem;                 // [256][33]
  float* Ss = Sf + 256 * 33;                // [256][33]
  __bf16* qb  = (__bf16*)(Ss + 256 * 33);   // [64][256]
  __bf16* kb  = qb + 64 * 256;              // [64][256]
  __bf16* wb  = kb + 64 * 256;              // [64][256]
  __bf16* Sb  = wb + 64 * 256;              // [256][32]
  __bf16* uib = Sb + 256 * 32;              // [64][32]
  __bf16* ab  = uib + 64 * 32;              // [64][64]

  int vb = blockIdx.x % VBLK;
  int bh = blockIdx.x / VBLK;
  int b = bh / HH, h = bh % HH;
  int tid = threadIdx.x, lane = tid & 31, wave = tid >> 5;
  int nn = lane & 15, gg = lane >> 4;
  float lf = lamf[h], ls = lams[h];

  for (int i = tid; i < 256 * 33; i += 256) { Sf[i] = 0.f; Ss[i] = 0.f; }
  __syncthreads();

  for (int n = 0; n < NCC; ++n) {
    float p = psig[(size_t)bh * NCC + n];
    size_t gbase = ((size_t)b * LL + (size_t)n * CC) * DD + (size_t)h * DKK;
    // staged bulk copy: 3 x 2048 uint4
    {
      uint4 rq[8], rk[8], rw8[8];
#pragma unroll
      for (int i = 0; i < 8; ++i) {
        int j = tid + i * 256;
        int c = j >> 5, c8 = j & 31;
        size_t off8 = (gbase + (size_t)c * DD) >> 3;
        rq[i]  = ((const uint4*)qgb)[off8 + c8];
        rk[i]  = ((const uint4*)kgb)[off8 + c8];
        rw8[i] = ((const uint4*)wgb)[off8 + c8];
      }
#pragma unroll
      for (int i = 0; i < 8; ++i) {
        int j = tid + i * 256;
        ((uint4*)qb)[j] = rq[i];
        ((uint4*)kb)[j] = rk[i];
        ((uint4*)wb)[j] = rw8[i];
      }
    }
    for (int i = tid; i < 256 * 32; i += 256) {
      int d = i >> 5, v = i & 31;
      Sb[i] = (__bf16)(Sf[d * 33 + v] + Ss[d * 33 + v]);
    }
    __syncthreads();

    // u_i = u - w @ S
    {
      int tm = wave >> 1, tn = wave & 1;
      v8f acc = {};
      for (int kk = 0; kk < 256; kk += 32) {
        v16bf a   = load_a_rm(wb + (tm * 16) * 256 + kk, 256);
        v16bf bfr = load_b_km(Sb + kk * 32 + tn * 16, 32);
        acc = wmma_bf(a, bfr, acc);
      }
#pragma unroll
      for (int r = 0; r < 8; ++r) {
        int c = tm * 16 + r + 8 * gg, v = tn * 16 + nn;
        float uv = ug[gbase + (size_t)c * DD + vb * VW + v] - acc[r];
        uib[c * 32 + v] = (__bf16)uv;
      }
    }
    // attn = (q @ k^T) * causal
#pragma unroll
    for (int t2 = 0; t2 < 2; ++t2) {
      int ti = wave * 2 + t2, tm = ti >> 2, tn = ti & 3;
      v8f acc = {};
      for (int kk = 0; kk < 256; kk += 32) {
        v16bf a   = load_a_rm(qb + (tm * 16) * 256 + kk, 256);
        v16bf bfr = load_b_nm(kb + (tn * 16) * 256 + kk, 256);
        acc = wmma_bf(a, bfr, acc);
      }
#pragma unroll
      for (int r = 0; r < 8; ++r) {
        int gm = tm * 16 + r + 8 * gg, gn = tn * 16 + nn;
        ab[gm * 64 + gn] = (__bf16)(gm >= gn ? acc[r] : 0.f);
      }
    }
    __syncthreads();

    // o = q @ S + attn @ u_i
    {
      int tm = wave >> 1, tn = wave & 1;
      v8f acc = {};
      for (int kk = 0; kk < 256; kk += 32) {
        v16bf a   = load_a_rm(qb + (tm * 16) * 256 + kk, 256);
        v16bf bfr = load_b_km(Sb + kk * 32 + tn * 16, 32);
        acc = wmma_bf(a, bfr, acc);
      }
      for (int kk = 0; kk < 64; kk += 32) {
        v16bf a   = load_a_rm(ab + (tm * 16) * 64 + kk, 64);
        v16bf bfr = load_b_km(uib + kk * 32 + tn * 16, 32);
        acc = wmma_bf(a, bfr, acc);
      }
#pragma unroll
      for (int r = 0; r < 8; ++r) {
        int c = tm * 16 + r + 8 * gg, v = tn * 16 + nn;
        og[gbase + (size_t)c * DD + vb * VW + v] = acc[r];
      }
    }
    // dS = k^T @ u_i ; dual-timescale state update
#pragma unroll
    for (int t4 = 0; t4 < 4; ++t4) {
      int ti = wave * 4 + t4, tm = ti >> 1, tn = ti & 1;
      v8f acc = {};
      for (int kk = 0; kk < 64; kk += 32) {
        v16bf a   = load_a_cm(kb + kk * 256 + tm * 16, 256);
        v16bf bfr = load_b_km(uib + kk * 32 + tn * 16, 32);
        acc = wmma_bf(a, bfr, acc);
      }
#pragma unroll
      for (int r = 0; r < 8; ++r) {
        int d = tm * 16 + r + 8 * gg, v = tn * 16 + nn;
        int si = d * 33 + v;
        float ds = acc[r];
        Sf[si] = lf * Sf[si] + p * ds;
        Ss[si] = ls * Ss[si] + (1.f - p) * ds;
      }
    }
    __syncthreads();
  }
}

// =====================================================================
// Kernel 6: RMS-norm + gate (bf16 g) -> bf16 (feeds final GEMM)
// =====================================================================
__global__ __launch_bounds__(128) void gate_kernel(const float* __restrict__ op,
                                                   const __bf16* __restrict__ gp,
                                                   const float* __restrict__ rmsw,
                                                   __bf16* __restrict__ out) {
  int row = blockIdx.x, lane = threadIdx.x & 31, h = threadIdx.x >> 5;
  size_t base = (size_t)row * DD + (size_t)h * DKK + lane * 8;
  float4 o0 = *(const float4*)(op + base);
  float4 o1 = *(const float4*)(op + base + 4);
  float ss = waveSum(dot4(o0, o0) + dot4(o1, o1));
  float inv = rsqrtf(ss * (1.f / 256.f) + 1e-5f);
  float4 g0, g1;
  unpack8(*(const uint4*)(gp + base), g0, g1);
  float4 r0 = *(const float4*)(rmsw + lane * 8);
  float4 r1 = *(const float4*)(rmsw + lane * 8 + 4);
  float4 y0 = scale4(mul4(mul4(o0, r0), sig4(g0)), inv);
  float4 y1 = scale4(mul4(mul4(o1, r1), sig4(g1)), inv);
  *(uint4*)(out + base) = pack8(y0, y1);
}

// =====================================================================
extern "C" void kernel_launch(void* const* d_in, const int* in_sizes, int n_in,
                              void* d_out, int out_size, void* d_ws, size_t ws_size,
                              hipStream_t stream) {
  const float* x    = (const float*)d_in[0];
  const float* Wq   = (const float*)d_in[1];
  const float* Wk   = (const float*)d_in[2];
  const float* Wv   = (const float*)d_in[3];
  const float* Wb   = (const float*)d_in[4];
  const float* Wg   = (const float*)d_in[5];
  const float* Wo   = (const float*)d_in[6];
  const float* cq   = (const float*)d_in[7];
  const float* ck   = (const float*)d_in[8];
  const float* cv   = (const float*)d_in[9];
  const float* Wbil = (const float*)d_in[10];
  const float* temp = (const float*)d_in[11];
  const float* fw1  = (const float*)d_in[12];
  const float* fb1  = (const float*)d_in[13];
  const float* fw2  = (const float*)d_in[14];
  const float* fb2  = (const float*)d_in[15];
  const float* rmsw = (const float*)d_in[16];
  const float* lamf = (const float*)d_in[17];
  const float* lams = (const float*)d_in[18];

  const size_t SZ = (size_t)BB * LL * DD;
  char* wsb = (char*)d_ws;
  auto alloc = [&](size_t bytes) {
    char* p = wsb;
    wsb += (bytes + 255) & ~(size_t)255;
    return p;
  };
  float*  U    = (float*)alloc(SZ * 4);          // u (f32)
  float*  Opre = (float*)alloc(SZ * 4);          // o_pre (f32)
  float*  Beta = (float*)alloc((size_t)BB * LL * HH * 4);
  float*  Psi  = (float*)alloc((size_t)BB * HH * NCC * 4);
  __bf16* xbf  = (__bf16*)alloc(SZ * 2);
  __bf16* P0   = (__bf16*)alloc(SZ * 2);         // xq -> g (reused)
  __bf16* P1   = (__bf16*)alloc(SZ * 2);         // xk
  __bf16* P2   = (__bf16*)alloc(SZ * 2);         // xv
  __bf16* QC   = (__bf16*)alloc(SZ * 2);         // conv q
  __bf16* KC   = (__bf16*)alloc(SZ * 2);         // conv k
  __bf16* VC   = (__bf16*)alloc(SZ * 2);         // conv v
  __bf16* qbfg = (__bf16*)alloc(SZ * 2);
  __bf16* kbfg = (__bf16*)alloc(SZ * 2);
  __bf16* wbfg = (__bf16*)alloc(SZ * 2);
  __bf16* obfg = (__bf16*)alloc(SZ * 2);
  __bf16* Wqt = (__bf16*)alloc((size_t)DD * DD * 2);
  __bf16* Wkt = (__bf16*)alloc((size_t)DD * DD * 2);
  __bf16* Wvt = (__bf16*)alloc((size_t)DD * DD * 2);
  __bf16* Wgt = (__bf16*)alloc((size_t)DD * DD * 2);
  __bf16* Wot = (__bf16*)alloc((size_t)DD * DD * 2);

  const int M = BB * LL, N = DD, K = DD;
  dim3 gemmGrid(M / GBM, N / GBN);
  dim3 trGrid(DD / 64, DD / 64);

  cvt_bf16_kernel<<<(int)(SZ / 8 / 256), 256, 0, stream>>>(x, xbf, SZ / 8);
  cvtT_bf16_kernel<<<trGrid, 256, 0, stream>>>(Wq, Wqt, K, N);
  cvtT_bf16_kernel<<<trGrid, 256, 0, stream>>>(Wk, Wkt, K, N);
  cvtT_bf16_kernel<<<trGrid, 256, 0, stream>>>(Wv, Wvt, K, N);
  cvtT_bf16_kernel<<<trGrid, 256, 0, stream>>>(Wg, Wgt, K, N);
  cvtT_bf16_kernel<<<trGrid, 256, 0, stream>>>(Wo, Wot, K, N);

  // projections (bf16 out)
  gemm_bf16_kernel<true><<<gemmGrid, 256, 0, stream>>>(xbf, Wqt, P0, M, N, K);
  gemm_bf16_kernel<true><<<gemmGrid, 256, 0, stream>>>(xbf, Wkt, P1, M, N, K);
  gemm_bf16_kernel<true><<<gemmGrid, 256, 0, stream>>>(xbf, Wvt, P2, M, N, K);
  beta_kernel<<<M / 8, 256, 0, stream>>>(x, Wb, Beta);

  int cblocks = (int)(SZ / 4 / 256);
  conv_silu_kernel<<<cblocks, 256, 0, stream>>>(P0, cq, QC);
  conv_silu_kernel<<<cblocks, 256, 0, stream>>>(P1, ck, KC);
  conv_silu_kernel<<<cblocks, 256, 0, stream>>>(P2, cv, VC);

  const size_t chunkLds =
      (size_t)(64 * 65 * 2 + 256 * 2 + 128 + 16) * 4 + (size_t)(4 * 64 * 256 + 64 * 64) * 2;
  chunk_kernel<<<BB * HH * NCC, 256, chunkLds, stream>>>(
      QC, KC, VC, Beta, Wbil, temp, fw1, fb1, fw2, fb2,
      qbfg, kbfg, wbfg, U, Psi);

  // gate projection (bf16 out, reuse P0)
  gemm_bf16_kernel<true><<<gemmGrid, 256, 0, stream>>>(xbf, Wgt, P0, M, N, K);

  const size_t scanLds =
      (size_t)(2 * 256 * 33) * 4 +
      (size_t)(3 * 64 * 256 + 256 * 32 + 64 * 32 + 64 * 64) * 2;
  scan_kernel<<<BB * HH * VBLK, 256, scanLds, stream>>>(
      qbfg, kbfg, wbfg, U, Psi, lamf, lams, Opre);

  gate_kernel<<<M, 128, 0, stream>>>(Opre, P0, rmsw, obfg);
  gemm_bf16_kernel<false><<<gemmGrid, 256, 0, stream>>>(obfg, Wot, (void*)d_out, M, N, K);
}
